// _ClassLinearAggBlock_81922206204652
// MI455X (gfx1250) — compile-verified, matrix operand order checked by reading, not run on hardware
//
#include <hip/hip_runtime.h>
#include <cstdint>

typedef __bf16 bf16;
typedef bf16  v16bf __attribute__((ext_vector_type(16)));
typedef float v8f   __attribute__((ext_vector_type(8)));

// ---- problem constants (from reference setup) ----
#define BB   4
#define TT   32
#define CC   128
#define HS   32
#define WS   32
#define GD   64
#define CG   192      // C + Gd
#define FF   512      // 4*C
#define NPIX 4096     // B*H*W
#define HWS  1024     // H*W

static __device__ __forceinline__ v8f vzero8() {
  v8f z;
#pragma unroll
  for (int i = 0; i < 8; ++i) z[i] = 0.0f;
  return z;
}

static __device__ __forceinline__ v8f wmma_bf16(v16bf a, v16bf b, v8f c) {
  return __builtin_amdgcn_wmma_f32_16x16x32_bf16(
      /*neg_a=*/false, a, /*neg_b=*/false, b,
      /*c_mod=*/(short)0, c, /*reuse_a=*/false, /*reuse_b=*/false);
}

// Load a 16x32 bf16 operand tile (A-layout; B uses the mirrored layout with
// pre-transposed NxK weights, so the same loader works for both).
// lanes 0-15: row rcBase+lane, K = {k..k+7, k+16..k+23}
// lanes 16-31: row rcBase+lane-16, K = {k+8..k+15, k+24..k+31}
template <typename P>
static __device__ __forceinline__ v16bf ld_tile(P base, int stride, int rcBase,
                                                int kBase, int lane) {
  int r = lane & 15, half = lane >> 4;
  P p = base + (size_t)(rcBase + r) * stride + kBase + half * 8;
  v16bf a;
#pragma unroll
  for (int i = 0; i < 8; ++i) { a[i] = p[i]; a[i + 8] = p[16 + i]; }
  return a;
}

// ---- weight prep: fp32 -> bf16, transposed to (N x K) row-major ----
// ws layout (bf16 elements): Wqt[0..24576) Wkt[..49152) Wvt[..65536)
//                            W1t[..131072) W2t[..196608)
__global__ void prep_weights(const float* __restrict__ Wq,
                             const float* __restrict__ Wk,
                             const float* __restrict__ Wv,
                             const float* __restrict__ W1,
                             const float* __restrict__ W2,
                             unsigned short* __restrict__ wsu) {
  bf16* o = reinterpret_cast<bf16*>(wsu);
  int i = blockIdx.x * blockDim.x + threadIdx.x;
  if (i >= 196608) return;
  int j = i;
  if (j < 24576) { int n = j / CG, k = j % CG; o[i] = (bf16)Wq[k * CC + n]; return; }
  j -= 24576;
  if (j < 24576) { int n = j / CG, k = j % CG; o[i] = (bf16)Wk[k * CC + n]; return; }
  j -= 24576;
  if (j < 16384) { int n = j / CC, k = j % CC; o[i] = (bf16)Wv[k * CC + n]; return; }
  j -= 16384;
  if (j < 65536) { int n = j / CC, k = j % CC; o[i] = (bf16)W1[k * FF + n]; return; }
  j -= 65536;
  { int n = j / FF, k = j % FF; o[i] = (bf16)W2[k * CC + n]; }
}

// ---- fused block: one workgroup per pixel n (B*H*W of them) ----
__global__ __launch_bounds__(128)
void fused_block(const float* __restrict__ x, const float* __restrict__ gdn,
                 const float* __restrict__ bq, const float* __restrict__ bk,
                 const float* __restrict__ bv,
                 const float* __restrict__ ln1g, const float* __restrict__ ln1b,
                 const float* __restrict__ ln2g, const float* __restrict__ ln2b,
                 const float* __restrict__ b1, const float* __restrict__ b2,
                 const unsigned short* __restrict__ wsu,
                 float* __restrict__ out) {
  __shared__ unsigned char smem[65536];
  // phase-reused LDS layout
  float* xLDS = (float*)(smem + 0);          // 32x128 f32   (dead after P3)
  bf16*  xqk  = (bf16*)(smem + 16384);       // 32x192 bf16  (dead after P1)
  bf16*  Qs   = (bf16*)(smem + 28672);       // 32x128 bf16  (dead after P3)
  bf16*  Kt   = (bf16*)(smem + 36864);       // 128x32 bf16  (dead after P2)
  bf16*  Vt   = (bf16*)(smem + 45056);       // 128x32 bf16  (dead after P2)
  bf16*  KVt  = (bf16*)(smem + 53248);       // 128x32 bf16  (dead after P3)
  float* Ksum = (float*)(smem + 61440);      // 128 f32
  float* Zl   = (float*)(smem + 61952);      // 128 f32 (l*4+h)
  float* mu1  = (float*)(smem + 62464);      // 32
  float* rv1  = (float*)(smem + 62592);      // 32
  float* mu2  = (float*)(smem + 62720);      // 32
  float* rv2  = (float*)(smem + 62848);      // 32
  float* out1 = (float*)(smem + 36864);      // 32x128 f32 (over Kt+Vt)
  bf16*  h0   = (bf16*)(smem + 53248);       // 32x128 bf16 (over KVt)
  bf16*  h1   = (bf16*)(smem + 0);           // 32x512 bf16 (over x,xqk,Q)

  const bf16* Wqt = reinterpret_cast<const bf16*>(wsu);
  const bf16* Wkt = Wqt + 24576;
  const bf16* Wvt = Wqt + 49152;
  const bf16* W1t = Wqt + 65536;
  const bf16* W2t = Wqt + 131072;

  const int tid  = threadIdx.x;
  const int lane = tid & 31;
  const int wv   = tid >> 5;          // wave id 0..3
  const int r    = lane & 15;
  const int half = lane >> 4;

  const int n   = blockIdx.x;
  const int b   = n >> 10;
  const int hs  = (n >> 5) & 31;
  const int wsp = n & 31;
  const float* xn = x + (size_t)b * TT * CC * HWS + hs * WS + wsp;

  // -------- P0: stage x and xqk=[x|guidance] --------
  for (int i = tid; i < TT * CC; i += 128) {
    int t = i >> 7, c = i & 127;
    float v = xn[(size_t)(t * CC + c) * HWS];
    xLDS[i] = v;
    xqk[t * CG + c] = (bf16)v;
  }
  for (int i = tid; i < TT * GD; i += 128) {
    int t = i >> 6, j = i & 63;
    xqk[t * CG + 128 + j] = (bf16)gdn[((size_t)b * TT + t) * GD + j];
  }
  __syncthreads();

  // -------- P1: q,k (K=192) and v (K=128) GEMMs --------
  {
    v8f aq[2][2], ak[2][2], av[2][2];
#pragma unroll
    for (int m = 0; m < 2; ++m)
#pragma unroll
      for (int j = 0; j < 2; ++j) { aq[m][j] = vzero8(); ak[m][j] = vzero8(); av[m][j] = vzero8(); }
#pragma unroll
    for (int kb = 0; kb < 6; ++kb) {
      v16bf A0 = ld_tile(xqk, CG, 0, kb * 32, lane);
      v16bf A1 = ld_tile(xqk, CG, 16, kb * 32, lane);
#pragma unroll
      for (int j = 0; j < 2; ++j) {
        int nt = 2 * wv + j;
        v16bf Bq = ld_tile(Wqt, CG, nt * 16, kb * 32, lane);
        aq[0][j] = wmma_bf16(A0, Bq, aq[0][j]);
        aq[1][j] = wmma_bf16(A1, Bq, aq[1][j]);
        v16bf Bk = ld_tile(Wkt, CG, nt * 16, kb * 32, lane);
        ak[0][j] = wmma_bf16(A0, Bk, ak[0][j]);
        ak[1][j] = wmma_bf16(A1, Bk, ak[1][j]);
        if (kb < 4) {
          v16bf Bv = ld_tile(Wvt, CC, nt * 16, kb * 32, lane);
          av[0][j] = wmma_bf16(A0, Bv, av[0][j]);
          av[1][j] = wmma_bf16(A1, Bv, av[1][j]);
        }
      }
    }
    // epilogue: bias, elu+1, scatter Q (row-major) and K^T, V^T to LDS
#pragma unroll
    for (int j = 0; j < 2; ++j) {
      int col = (2 * wv + j) * 16 + r;
      float bqv = bq[col], bkv = bk[col], bvv = bv[col];
#pragma unroll
      for (int mt = 0; mt < 2; ++mt)
#pragma unroll
        for (int i = 0; i < 8; ++i) {
          int l = mt * 16 + half * 8 + i;
          float qv = aq[mt][j][i] + bqv;
          qv = (qv > 0.f) ? (qv + 1.f) : __expf(qv);
          Qs[l * CC + col] = (bf16)qv;
          float kv = ak[mt][j][i] + bkv;
          kv = (kv > 0.f) ? (kv + 1.f) : __expf(kv);
          Kt[col * TT + l] = (bf16)kv;
          Vt[col * TT + l] = (bf16)(av[mt][j][i] + bvv);
        }
    }
  }
  __syncthreads();

  // -------- P2: Ksum, KV = K^T @ V (per head, one WMMA K-step) --------
  {
    float s = 0.f;
#pragma unroll
    for (int i = 0; i < TT; ++i) s += (float)Kt[tid * TT + i];
    Ksum[tid] = s;
  }
  {
    int h = wv;
    v8f acc[2][2];
#pragma unroll
    for (int m = 0; m < 2; ++m)
#pragma unroll
      for (int j = 0; j < 2; ++j) acc[m][j] = vzero8();
#pragma unroll
    for (int mt = 0; mt < 2; ++mt) {
      v16bf A = ld_tile(Kt, TT, h * 32 + mt * 16, 0, lane);
#pragma unroll
      for (int nt = 0; nt < 2; ++nt) {
        v16bf Bv = ld_tile(Vt, TT, h * 32 + nt * 16, 0, lane);
        acc[mt][nt] = wmma_bf16(A, Bv, acc[mt][nt]);
      }
    }
#pragma unroll
    for (int mt = 0; mt < 2; ++mt)
#pragma unroll
      for (int nt = 0; nt < 2; ++nt)
#pragma unroll
        for (int i = 0; i < 8; ++i) {
          int d = mt * 16 + half * 8 + i;       // row of KV
          int e = h * 32 + nt * 16 + r;          // global v-column
          KVt[e * 32 + d] = (bf16)acc[mt][nt][i];
        }
  }
  __syncthreads();

  // -------- P2c: Z = 1/(Q.Ksum+eps), ln1 stats --------
  {
    int l = tid >> 2, h = tid & 3;
    float d = 0.f;
#pragma unroll
    for (int i = 0; i < 32; ++i)
      d += (float)Qs[l * CC + h * 32 + i] * Ksum[h * 32 + i];
    Zl[tid] = 1.0f / (d + 1e-6f);
  }
  if (tid < 32) {
    float s = 0.f;
    for (int c = 0; c < CC; ++c) s += xLDS[tid * CC + c];
    float mu = s * (1.0f / CC);
    float v2 = 0.f;
    for (int c = 0; c < CC; ++c) { float d = xLDS[tid * CC + c] - mu; v2 += d * d; }
    mu1[tid] = mu;
    rv1[tid] = rsqrtf(v2 * (1.0f / CC) + 1e-5f);
  }
  __syncthreads();

  // -------- P3: out = (Q @ KV)*Z + LN1(x)  (writes out1 over Kt/Vt) --------
  {
    int h = wv;
    v8f acc[2][2];
#pragma unroll
    for (int m = 0; m < 2; ++m)
#pragma unroll
      for (int j = 0; j < 2; ++j) acc[m][j] = vzero8();
#pragma unroll
    for (int mt = 0; mt < 2; ++mt) {
      v16bf A = ld_tile(Qs, CC, mt * 16, h * 32, lane);
#pragma unroll
      for (int nt = 0; nt < 2; ++nt) {
        v16bf Bv = ld_tile(KVt, 32, h * 32 + nt * 16, 0, lane);
        acc[mt][nt] = wmma_bf16(A, Bv, acc[mt][nt]);
      }
    }
#pragma unroll
    for (int mt = 0; mt < 2; ++mt)
#pragma unroll
      for (int nt = 0; nt < 2; ++nt)
#pragma unroll
        for (int i = 0; i < 8; ++i) {
          int l = mt * 16 + half * 8 + i;
          int c = h * 32 + nt * 16 + r;
          float res = acc[mt][nt][i] * Zl[l * 4 + h];
          res += (xLDS[l * CC + c] - mu1[l]) * rv1[l] * ln1g[c] + ln1b[c];
          out1[l * CC + c] = res;
        }
  }
  __syncthreads();

  // -------- P4: LN2 -> h0 (bf16) --------
  if (tid < 32) {
    float s = 0.f;
    for (int c = 0; c < CC; ++c) s += out1[tid * CC + c];
    float mu = s * (1.0f / CC);
    float v2 = 0.f;
    for (int c = 0; c < CC; ++c) { float d = out1[tid * CC + c] - mu; v2 += d * d; }
    mu2[tid] = mu;
    rv2[tid] = rsqrtf(v2 * (1.0f / CC) + 1e-5f);
  }
  __syncthreads();
  for (int i = tid; i < TT * CC; i += 128) {
    int l = i >> 7, c = i & 127;
    h0[i] = (bf16)((out1[i] - mu2[l]) * rv2[l] * ln2g[c] + ln2b[c]);
  }
  __syncthreads();

  // -------- P5: h1 = gelu(h0 @ W1 + b1)  (32x512, over x/xqk/Q) --------
  {
    v16bf Ah[2][4];
#pragma unroll
    for (int mt = 0; mt < 2; ++mt)
#pragma unroll
      for (int kb = 0; kb < 4; ++kb)
        Ah[mt][kb] = ld_tile(h0, CC, mt * 16, kb * 32, lane);
#pragma unroll
    for (int jn = 0; jn < 8; ++jn) {
      int nt = wv * 8 + jn;
      v8f acc[2] = {vzero8(), vzero8()};
#pragma unroll
      for (int kb = 0; kb < 4; ++kb) {
        v16bf Bv = ld_tile(W1t, CC, nt * 16, kb * 32, lane);
        acc[0] = wmma_bf16(Ah[0][kb], Bv, acc[0]);
        acc[1] = wmma_bf16(Ah[1][kb], Bv, acc[1]);
      }
      int col = nt * 16 + r;
      float bb = b1[col];
#pragma unroll
      for (int mt = 0; mt < 2; ++mt)
#pragma unroll
        for (int i = 0; i < 8; ++i) {
          int l = mt * 16 + half * 8 + i;
          float hv = acc[mt][i] + bb;
          hv = 0.5f * hv * (1.0f + erff(hv * 0.70710678f));  // exact gelu
          h1[l * FF + col] = (bf16)hv;
        }
    }
  }
  __syncthreads();

  // -------- P6: out = out1 + h1 @ W2 + b2, scatter to (B,T,C,H,W) --------
  {
    v8f acc[2][2];
#pragma unroll
    for (int m = 0; m < 2; ++m)
#pragma unroll
      for (int j = 0; j < 2; ++j) acc[m][j] = vzero8();
#pragma unroll
    for (int kb = 0; kb < 16; ++kb) {
      v16bf A0 = ld_tile(h1, FF, 0, kb * 32, lane);
      v16bf A1 = ld_tile(h1, FF, 16, kb * 32, lane);
#pragma unroll
      for (int j = 0; j < 2; ++j) {
        int nt = 2 * wv + j;
        v16bf Bv = ld_tile(W2t, FF, nt * 16, kb * 32, lane);
        acc[0][j] = wmma_bf16(A0, Bv, acc[0][j]);
        acc[1][j] = wmma_bf16(A1, Bv, acc[1][j]);
      }
    }
#pragma unroll
    for (int j = 0; j < 2; ++j) {
      int c = (2 * wv + j) * 16 + r;
      float b2v = b2[c];
#pragma unroll
      for (int mt = 0; mt < 2; ++mt)
#pragma unroll
        for (int i = 0; i < 8; ++i) {
          int l = mt * 16 + half * 8 + i;
          float val = acc[mt][j][i] + out1[l * CC + c] + b2v;
          out[((size_t)(b * TT + l) * CC + c) * HWS + hs * WS + wsp] = val;
        }
    }
  }
}

extern "C" void kernel_launch(void* const* d_in, const int* in_sizes, int n_in,
                              void* d_out, int out_size, void* d_ws, size_t ws_size,
                              hipStream_t stream) {
  (void)in_sizes; (void)n_in; (void)out_size; (void)ws_size;
  const float* x    = (const float*)d_in[0];
  const float* g    = (const float*)d_in[1];
  const float* Wq   = (const float*)d_in[2];
  const float* bq   = (const float*)d_in[3];
  const float* Wk   = (const float*)d_in[4];
  const float* bk   = (const float*)d_in[5];
  const float* Wv   = (const float*)d_in[6];
  const float* bv   = (const float*)d_in[7];
  const float* ln1g = (const float*)d_in[8];
  const float* ln1b = (const float*)d_in[9];
  const float* ln2g = (const float*)d_in[10];
  const float* ln2b = (const float*)d_in[11];
  const float* W1   = (const float*)d_in[12];
  const float* b1   = (const float*)d_in[13];
  const float* W2   = (const float*)d_in[14];
  const float* b2   = (const float*)d_in[15];
  unsigned short* wt = (unsigned short*)d_ws;   // 196608 bf16 = 384 KB

  prep_weights<<<(196608 + 255) / 256, 256, 0, stream>>>(Wq, Wk, Wv, W1, W2, wt);
  fused_block<<<NPIX, 128, 0, stream>>>(x, g, bq, bk, bv, ln1g, ln1b, ln2g, ln2b,
                                        b1, b2, wt, (float*)d_out);
}
